// RGCN_29815662968876
// MI455X (gfx1250) — matrix operand order repaired and checked by reading, not compiled, not run on hardware
//
#include <hip/hip_runtime.h>
#include <hip/hip_bf16.h>

typedef __attribute__((ext_vector_type(16))) _Float16 v16h;
typedef __attribute__((ext_vector_type(8)))  _Float16 v8h;
typedef __attribute__((ext_vector_type(2)))  _Float16 h2v;
typedef __attribute__((ext_vector_type(8)))  float    v8f;
typedef __attribute__((ext_vector_type(4)))  unsigned v4u;
typedef __attribute__((ext_vector_type(8)))  int      v8i;
typedef __attribute__((ext_vector_type(4)))  int      v4i;

__device__ __forceinline__ float eluf(float x) {
    return x > 0.0f ? x : (expf(x) - 1.0f);
}

// ---------------- counts ----------------
__global__ void rgcn_zero_u32(unsigned* __restrict__ p, int n) {
    int i = blockIdx.x * blockDim.x + threadIdx.x;
    if (i < n) p[i] = 0u;
}

__global__ void rgcn_count(const int* __restrict__ dst, const int* __restrict__ et,
                           unsigned* __restrict__ cnt, int E, int R) {
    int e = blockIdx.x * blockDim.x + threadIdx.x;
    if (e >= E) return;
    atomicAdd(&cnt[(size_t)dst[e] * R + et[e]], 1u);
}

// ---------------- conv1 ----------------
__global__ void rgcn_init_h1(const float* __restrict__ root1, const float* __restrict__ bias1,
                             float* __restrict__ h1, int total) {
    int i = blockIdx.x * blockDim.x + threadIdx.x;
    if (i < total) h1[i] = root1[i] + bias1[i & 63];
}

__global__ void rgcn_scatter1(const int* __restrict__ src, const int* __restrict__ dst,
                              const int* __restrict__ et, const unsigned* __restrict__ cnt,
                              const float* __restrict__ w1, float* __restrict__ h1,
                              int E, int N, int R) {
    int t = blockIdx.x * blockDim.x + threadIdx.x;
    int e = t >> 5, lane = t & 31;
    if (e >= E) return;
    int s = src[e], d = dst[e], r = et[e];
    float sc = 1.0f / fmaxf((float)cnt[(size_t)d * R + r], 1.0f);
    const float2* wrow = (const float2*)(w1 + ((size_t)r * N + (size_t)s) * 64);
    float2 v = wrow[lane];
    float* o = h1 + (size_t)d * 64 + lane * 2;
    atomicAdd(o,     v.x * sc);
    atomicAdd(o + 1, v.y * sc);
}

__global__ void rgcn_elu_h1(float* __restrict__ h1, _Float16* __restrict__ h1h, int total) {
    int i = blockIdx.x * blockDim.x + threadIdx.x;
    if (i >= total) return;
    float y = eluf(h1[i]);
    h1[i]  = y;
    h1h[i] = (_Float16)y;
}

__global__ void rgcn_cvt_w(const float* __restrict__ w2, const float* __restrict__ root2,
                           _Float16* __restrict__ w2h, int R) {
    int i = blockIdx.x * blockDim.x + threadIdx.x;
    int total = (R + 1) * 64 * 64;
    if (i >= total) return;
    int r = i >> 12;
    float v = (r < R) ? w2[i] : root2[i - (size_t)R * 4096];
    w2h[i] = (_Float16)v;
}

// ---------------- dense transforms via WMMA + TDM ----------------
// grid.y = r in [0..R]; r<R -> hr[r] (f16), r==R -> h2 = h1@root2 + bias2 (f32)
// Per block: TDM DMA of the 64x64 f16 weight tile into LDS, then 8 waves
// (4 col tiles x 2 node groups), each wave holding its packed B operand in
// registers and streaming GEMM_TPW node tiles (2 x v_wmma_f32_16x16x32_f16).
#define GEMM_TPW 16

__global__ void rgcn_gemm(const _Float16* __restrict__ h1h,
                          const _Float16* __restrict__ w2h,   // [(R+1)][64][64]
                          const float* __restrict__ bias2,
                          _Float16* __restrict__ hr,          // [R][N][64]
                          float* __restrict__ h2,             // [N][64]
                          int N, int R) {
    __shared__ _Float16 lds_w[64 * 64];

    int tid  = threadIdx.x;
    int wave = tid >> 5;
    int lane = tid & 31;
    int r    = blockIdx.y;

    // --- Tensor Data Mover: DMA w2h[r] (2D tile 64x64, 2B elems) into LDS ---
    if (wave == 0) {
        unsigned long long ga = (unsigned long long)(const void*)(w2h + (size_t)r * 4096);
        unsigned lds_off = (unsigned)(unsigned long long)(void*)&lds_w[0]; // LDS byte offset
        // D# group0: count=1 | lds_addr | global_addr[56:0] | type=2
        v4u g0 = { 1u, lds_off, (unsigned)ga,
                   ((unsigned)(ga >> 32) & 0x01FFFFFFu) | 0x80000000u };
        // D# group1: data_size=1(2B); tensor_dim0=64; tensor_dim1=64;
        //            tile_dim0=64; tile_dim1=64; tensor_dim0_stride=64
        v8i g1 = { (int)(1u << 16),   // data_size @ bits[17:16]
                   (int)(64u << 16),  // tensor_dim0 low16 @ bits[63:48]
                   (int)(64u << 16),  // tensor_dim0 hi=0, tensor_dim1 low16 @ [95:80]
                   (int)(64u << 16),  // tensor_dim1 hi=0, tile_dim0 @ [127:112]
                   64,                // tile_dim1 @ [143:128], tile_dim2=0
                   64,                // tensor_dim0_stride low32
                   0, 0 };
        v4i gz4 = { 0, 0, 0, 0 };
        v8i gz8 = { 0, 0, 0, 0, 0, 0, 0, 0 };
        __builtin_amdgcn_tensor_load_to_lds(g0, g1, gz4, gz4, gz8, 0);
        __builtin_amdgcn_s_wait_tensorcnt(0);
    }
    __syncthreads();

    int m = lane & 15;          // A row within tile / D column
    int g = lane >> 4;          // lane group
    int colTile = wave & 3;
    int sub     = wave >> 2;
    int col0 = colTile * 16;

    // Pack both K-step B operands once per wave from LDS.
    v16h B0, B1;
#pragma unroll
    for (int j = 0; j < 16; ++j) {
        int kk = (j < 8) ? (g * 8 + j) : (16 + g * 8 + (j - 8));
        B0[j] = lds_w[kk * 64 + col0 + m];
        B1[j] = lds_w[(32 + kk) * 64 + col0 + m];
    }

    int nTiles = N >> 4;        // N multiple of 16
    int base = blockIdx.x * (2 * GEMM_TPW) + sub * GEMM_TPW;

    for (int t = 0; t < GEMM_TPW; ++t) {
        int nodeTile = base + t;
        if (nodeTile >= nTiles) break;      // wave-uniform (EXEC stays all-1)

        const _Float16* arow = h1h + ((size_t)nodeTile * 16 + m) * 64;
        union { v16h v; v8h h[2]; } A0, A1;
        A0.h[0] = *(const v8h*)(arow + g * 8);
        A0.h[1] = *(const v8h*)(arow + 16 + g * 8);
        A1.h[0] = *(const v8h*)(arow + 32 + g * 8);
        A1.h[1] = *(const v8h*)(arow + 48 + g * 8);

        v8f acc = {};
        acc = __builtin_amdgcn_wmma_f32_16x16x32_f16(false, A0.v, false, B0,
                                                     (short)0, acc, false, false);
        acc = __builtin_amdgcn_wmma_f32_16x16x32_f16(false, A1.v, false, B1,
                                                     (short)0, acc, false, false);

        if (r < R) {
            _Float16* drow = hr + ((size_t)r * N + (size_t)nodeTile * 16) * 64 + col0 + m;
#pragma unroll
            for (int v = 0; v < 8; ++v)
                drow[(size_t)(g * 8 + v) * 64] = (_Float16)acc[v];
        } else {
            float* drow = h2 + (size_t)nodeTile * 16 * 64 + col0 + m;
            float b = bias2[col0 + m];
#pragma unroll
            for (int v = 0; v < 8; ++v)
                drow[(size_t)(g * 8 + v) * 64] = acc[v] + b;
        }
    }
}

// ---------------- conv2 scatter ----------------
__global__ void rgcn_scatter2(const int* __restrict__ src, const int* __restrict__ dst,
                              const int* __restrict__ et, const unsigned* __restrict__ cnt,
                              const _Float16* __restrict__ hr, float* __restrict__ h2,
                              int E, int N, int R) {
    int t = blockIdx.x * blockDim.x + threadIdx.x;
    int e = t >> 5, lane = t & 31;
    if (e >= E) return;
    int s = src[e], d = dst[e], r = et[e];
    float sc = 1.0f / fmaxf((float)cnt[(size_t)d * R + r], 1.0f);
    h2v v = *(const h2v*)(hr + ((size_t)r * N + (size_t)s) * 64 + lane * 2);
    float* o = h2 + (size_t)d * 64 + lane * 2;
    atomicAdd(o,     (float)v.x * sc);
    atomicAdd(o + 1, (float)v.y * sc);
}

// ---------------- head: out = elu(h2[:T]) @ lin_w + lin_b ----------------
__global__ void rgcn_head(const float* __restrict__ h2,
                          const float* __restrict__ lin_w,   // [64][16]
                          const float* __restrict__ lin_b,   // [16]
                          float* __restrict__ out, int T) {
    int wave = threadIdx.x >> 5;
    int lane = threadIdx.x & 31;
    int nTiles = (T + 15) >> 4;
    int rowTile = blockIdx.x * 8 + wave;
    if (rowTile >= nTiles) return;             // wave-uniform

    int m = lane & 15;
    int g = lane >> 4;
    int row = rowTile * 16 + m;
    if (row >= T) row = T - 1;                 // lane-level select, no EXEC change

    const float* arow = h2 + (size_t)row * 64;
    v8f acc = {};
#pragma unroll
    for (int s = 0; s < 2; ++s) {
        int k0 = s * 32;
        v16h A;
#pragma unroll
        for (int j = 0; j < 8; ++j) {
            A[j]     = (_Float16)eluf(arow[k0 + g * 8 + j]);
            A[j + 8] = (_Float16)eluf(arow[k0 + 16 + g * 8 + j]);
        }
        v16h B;
#pragma unroll
        for (int j = 0; j < 8; ++j) {
            B[j]     = (_Float16)lin_w[(size_t)(k0 + g * 8 + j) * 16 + m];
            B[j + 8] = (_Float16)lin_w[(size_t)(k0 + 16 + g * 8 + j) * 16 + m];
        }
        acc = __builtin_amdgcn_wmma_f32_16x16x32_f16(false, A, false, B,
                                                     (short)0, acc, false, false);
    }
    float b = lin_b[m];
#pragma unroll
    for (int v = 0; v < 8; ++v) {
        int rr = rowTile * 16 + g * 8 + v;
        if (rr < T) out[(size_t)rr * 16 + m] = acc[v] + b;
    }
}

static inline size_t alignup(size_t x) { return (x + 255) & ~(size_t)255; }

extern "C" void kernel_launch(void* const* d_in, const int* in_sizes, int n_in,
                              void* d_out, int out_size, void* d_ws, size_t ws_size,
                              hipStream_t stream) {
    const int*   edge_index = (const int*)d_in[0];
    const int*   edge_type  = (const int*)d_in[1];
    const float* weight1    = (const float*)d_in[3];
    const float* root1      = (const float*)d_in[4];
    const float* bias1      = (const float*)d_in[5];
    const float* weight2    = (const float*)d_in[6];
    const float* root2      = (const float*)d_in[7];
    const float* bias2      = (const float*)d_in[8];
    const float* lin_w      = (const float*)d_in[9];
    const float* lin_b      = (const float*)d_in[10];
    float*       out        = (float*)d_out;

    const int H = 64;
    const int C = in_sizes[10];                 // 16
    const int E = in_sizes[1];                  // 1.6M
    const int N = in_sizes[4] / H;              // 50000
    const int R = in_sizes[6] / (H * H);        // 8
    const int T = out_size / C;                 // 8192

    const int* src = edge_index;
    const int* dst = edge_index + E;

    // workspace layout
    char* ws = (char*)d_ws;
    size_t off = 0;
    unsigned*  cnt = (unsigned*)(ws + off);  off = alignup(off + (size_t)N * R * 4);
    float*     h1  = (float*)(ws + off);     off = alignup(off + (size_t)N * H * 4);
    _Float16*  h1h = (_Float16*)(ws + off);  off = alignup(off + (size_t)N * H * 2);
    float*     h2  = (float*)(ws + off);     off = alignup(off + (size_t)N * H * 4);
    _Float16*  w2h = (_Float16*)(ws + off);  off = alignup(off + (size_t)(R + 1) * H * H * 2);
    _Float16*  hr  = (_Float16*)(ws + off);  off = alignup(off + (size_t)R * N * H * 2);
    (void)ws_size;

    const int B256 = 256;

    // 1) per-(dst,relation) edge counts
    rgcn_zero_u32<<<(N * R + B256 - 1) / B256, B256, 0, stream>>>(cnt, N * R);
    rgcn_count<<<(E + B256 - 1) / B256, B256, 0, stream>>>(dst, edge_type, cnt, E, R);

    // 2) conv1: init + scatter-mean of embedding rows
    rgcn_init_h1<<<(N * H + B256 - 1) / B256, B256, 0, stream>>>(root1, bias1, h1, N * H);
    {
        long long threads = (long long)E * 32;
        rgcn_scatter1<<<(unsigned)((threads + B256 - 1) / B256), B256, 0, stream>>>(
            src, dst, edge_type, cnt, weight1, h1, E, N, R);
    }
    rgcn_elu_h1<<<(N * H + B256 - 1) / B256, B256, 0, stream>>>(h1, h1h, N * H);

    // 3) dense per-relation transforms + h2 init (TDM -> LDS -> WMMA)
    rgcn_cvt_w<<<((R + 1) * H * H + B256 - 1) / B256, B256, 0, stream>>>(weight2, root2, w2h, R);
    {
        int nTiles = N >> 4;
        int nodeBlocks = (nTiles + 2 * GEMM_TPW - 1) / (2 * GEMM_TPW);
        dim3 grid(nodeBlocks, R + 1);
        rgcn_gemm<<<grid, B256, 0, stream>>>(h1h, w2h, bias2, hr, h2, N, R);
    }

    // 4) conv2 scatter-mean from f16 hr
    {
        long long threads = (long long)E * 32;
        rgcn_scatter2<<<(unsigned)((threads + B256 - 1) / B256), B256, 0, stream>>>(
            src, dst, edge_type, cnt, hr, h2, E, N, R);
    }

    // 5) head (ELU fused into A-load, WMMA)
    {
        int nTiles = (T + 15) >> 4;
        rgcn_head<<<(nTiles + 7) / 8, B256, 0, stream>>>(h2, lin_w, lin_b, out, T);
    }
}
